// EideticIndexedLinearLayer_73469710565779
// MI455X (gfx1250) — matrix operand order repaired and checked by reading, not compiled, not run on hardware
//
#include <hip/hip_runtime.h>
#include <stddef.h>
#include <stdint.h>

#define B_DIM   2048
#define IN_DIM  1024
#define OUT_DIM 1024
#define Q_DIM   16

typedef __attribute__((ext_vector_type(16))) __bf16          v16bf;
typedef __attribute__((ext_vector_type(16))) unsigned short  v16u;
typedef __attribute__((ext_vector_type(8)))  float           v8f;

#if defined(__has_builtin)
#if __has_builtin(__builtin_amdgcn_global_load_async_to_lds_b128) && \
    __has_builtin(__builtin_amdgcn_s_wait_asynccnt)
#define EIDETIC_ASYNC_LDS 1
#endif
#endif

#ifdef EIDETIC_ASYNC_LDS
// Builtin signature (from compiler diagnostic): param is a pointer to a
// 4-int vector in the global address space; LDS side is AS(3).
typedef int v4i_gcc __attribute__((vector_size(16)));
typedef __attribute__((address_space(1))) v4i_gcc* as1_v4i_p;
typedef __attribute__((address_space(3))) v4i_gcc* as3_v4i_p;

// Per-lane 16 B global -> LDS copy on the async path (ASYNCcnt-tracked,
// no VGPR roundtrip). AS pointers built via uintptr_t: global flat address
// == AS1 address; flat LDS address low 32 bits == LDS offset (ISA 10.2).
__device__ __forceinline__ void async_b128(const void* g, void* l) {
  __builtin_amdgcn_global_load_async_to_lds_b128(
      (as1_v4i_p)(uintptr_t)g, (as3_v4i_p)(uintptr_t)l, 0, 0);
}
#endif

__device__ __forceinline__ unsigned short f32_to_bf16_rne(float f) {
  union { float f; unsigned int u; } cv; cv.f = f;
  unsigned int u = cv.u;
  unsigned int r = u + 0x7FFFu + ((u >> 16) & 1u);  // round-to-nearest-even
  return (unsigned short)(r >> 16);
}

// ---------------------------------------------------------------------------
// Pass 1: pack param_table [Q=16, IN, OUT] f32 -> bf16 in WMMA B-fragment
// order. Expanded contraction k' = k*16 + q; one K=32 WMMA slice covers
// features {2ks, 2ks+1} x quantiles {0..15}. Block per (ks, nt): 32 lanes x
// 16 bf16, lane l<16 -> (k=2ks, n=nt*16+l), lane l>=16 -> (k=2ks+1,
// n=nt*16+l-16), element e = quantile. Contiguous k order, same as the A fragment
// (lanes 0-15: K=0-15, lanes 16-31: K=16-31).
// ---------------------------------------------------------------------------
__global__ __launch_bounds__(256) void eidetic_pack_bp(
    const float* __restrict__ param, unsigned short* __restrict__ bp) {
  const unsigned tid = blockIdx.x * 256u + threadIdx.x;  // 2^20 threads
  const unsigned l   = tid & 31u;
  const unsigned nt  = (tid >> 5) & 63u;
  const unsigned ks  = tid >> 11;                        // 0..511
  const unsigned k   = 2u * ks + (l >> 4);
  const unsigned n   = nt * 16u + (l & 15u);
  const float* src = param + (size_t)k * OUT_DIM + n;
  v16u v;
#pragma unroll
  for (int e = 0; e < 16; ++e)  // e = quantile; q-stride = IN*OUT
    v[e] = f32_to_bf16_rne(src[(size_t)e * (IN_DIM * OUT_DIM)]);
  *reinterpret_cast<v16u*>(bp + (size_t)tid * 16u) = v;  // 2x global_store_b128
}

// Build one 16x32 bf16 A fragment in registers: lane handles row b; elems 0-15
// = one-hot(q) * x[b,kf] for q = 0..15, kf = k2 for lanes 0-15 (qb = 0) and
// k2+1 for lanes 16-31 (qb = 8); contiguous k order matching packed B.
__device__ __forceinline__ v16bf build_a_frag(const float* __restrict__ xr,
                                              const int*   __restrict__ ir,
                                              int k2, int qb) {
  float2 xv = *reinterpret_cast<const float2*>(xr + k2);
  int2   iv = *reinterpret_cast<const int2*>(ir + k2);
  unsigned short x0 = f32_to_bf16_rne(xv.x);
  unsigned short x1 = f32_to_bf16_rne(xv.y);
  v16u t;
#pragma unroll
  for (int e = 0; e < 16; ++e) {
    t[e] = ((qb ? iv.y : iv.x) == e) ? (qb ? x1 : x0) : (unsigned short)0;
  }
  return __builtin_bit_cast(v16bf, t);
}

// ---------------------------------------------------------------------------
// Pass 2: out = A' @ B' + bias.
// WG = 8 waves (4 wm x 2 wn), WG tile 128(m) x 128(n); wave tile 32 x 64
// (2 A-frags x 4 B-frags -> 8 WMMAs / K-step, 512 K-steps).
// Per K-step the WG consumes one contiguous 8 KB chunk of Bp, staged in
// double-buffered LDS one step ahead: async global->LDS copies when the
// toolchain exposes them (one barrier + s_wait_asynccnt per step), else
// global->reg->ds_store (two barriers per step). 4x reuse across wm waves.
// ---------------------------------------------------------------------------
__global__ __launch_bounds__(256) void eidetic_wmma_gemm(
    const float* __restrict__ x, const int* __restrict__ idx,
    const unsigned short* __restrict__ bp, const float* __restrict__ bias,
    float* __restrict__ out) {
  __shared__ alignas(16) unsigned short sB[2][8 * 512];  // 2 x 8 KB

  const int t    = threadIdx.x;
  const int lane = t & 31;
  const int w    = t >> 5;            // wave 0..7
  const int wm   = w >> 1;            // 0..3
  const int wn   = w & 1;             // 0..1
  const int m0   = blockIdx.y * 128 + wm * 32;  // batch-row base
  const int ntw  = wn * 4;                      // wave's local n-tile base
  const int ntwg = blockIdx.x * 8;              // WG's global n-tile base

  const int l  = lane & 15;
  const int hi = lane >> 4;
  const int qb = hi * 8;

  const float* xr0 = x   + (size_t)(m0 + l)      * IN_DIM;
  const float* xr1 = x   + (size_t)(m0 + 16 + l) * IN_DIM;
  const int*   ir0 = idx + (size_t)(m0 + l)      * IN_DIM;
  const int*   ir1 = idx + (size_t)(m0 + 16 + l) * IN_DIM;

  // This thread's 32-byte slice of the WG's per-step 8 KB weight chunk.
  // Chunk for step ks starts at Bp element (ks*64 + ntwg)*512.
  const unsigned short* bsrc = bp + (size_t)ntwg * 512 + (size_t)t * 16;
  const size_t kstride = 64 * 512;  // ushorts per K-step across all 64 n-tiles

  v8f c[2][4];
#pragma unroll
  for (int i = 0; i < 2; ++i)
#pragma unroll
    for (int j = 0; j < 4; ++j)
#pragma unroll
      for (int r = 0; r < 8; ++r) c[i][j][r] = 0.0f;

#ifdef EIDETIC_ASYNC_LDS
  // Prime: chunk 0 straight into sB[0] via async DMA.
  async_b128(bsrc,     &sB[0][t * 16]);
  async_b128(bsrc + 8, &sB[0][t * 16 + 8]);
  __builtin_amdgcn_s_wait_asynccnt(0);
  __syncthreads();
#else
  // Prime: chunk 0 into registers.
  uint4 st0 = *reinterpret_cast<const uint4*>(bsrc);
  uint4 st1 = *reinterpret_cast<const uint4*>(bsrc + 8);
#endif

  for (int ks = 0; ks < 512; ++ks) {
    const int buf = ks & 1;
    const unsigned short* nxt = bsrc + (size_t)((ks + 1) & 511) * kstride;

#ifdef EIDETIC_ASYNC_LDS
    // Issue next step's chunk into the other buffer; its previous readers
    // finished before the trailing barrier of the previous iteration.
    async_b128(nxt,     &sB[buf ^ 1][t * 16]);
    async_b128(nxt + 8, &sB[buf ^ 1][t * 16 + 8]);
#else
    // Commit staged chunk (data for this step) to LDS, then issue next loads.
    *reinterpret_cast<uint4*>(&sB[buf][t * 16])     = st0;  // ds_store_b128
    *reinterpret_cast<uint4*>(&sB[buf][t * 16 + 8]) = st1;
    st0 = *reinterpret_cast<const uint4*>(nxt);
    st1 = *reinterpret_cast<const uint4*>(nxt + 8);
    __syncthreads();  // staged data visible to all waves
#endif
    __builtin_prefetch(bsrc + (size_t)((ks + 2) & 511) * kstride, 0, 3);

    // A fragments from registers (one-hot routing of x by quantile index).
    const int k2 = ks * 2;
    v16bf a0 = build_a_frag(xr0, ir0, k2, qb);
    v16bf a1 = build_a_frag(xr1, ir1, k2, qb);

    // B fragments from LDS (lane-contiguous 32 B -> 2x ds_load_b128).
    v16bf b[4];
#pragma unroll
    for (int j = 0; j < 4; ++j)
      b[j] = __builtin_bit_cast(
          v16bf, *reinterpret_cast<const v16u*>(&sB[buf][(ntw + j) * 512 + lane * 16]));

#pragma unroll
    for (int j = 0; j < 4; ++j) {
      c[0][j] = __builtin_amdgcn_wmma_f32_16x16x32_bf16(
          false, a0, false, b[j], (short)0, c[0][j], false, false);
      c[1][j] = __builtin_amdgcn_wmma_f32_16x16x32_bf16(
          false, a1, false, b[j], (short)0, c[1][j], false, false);
    }

#ifdef EIDETIC_ASYNC_LDS
    __builtin_amdgcn_s_wait_asynccnt(0);  // next buffer fully written
#endif
    __syncthreads();  // all waves done reading sB[buf] / next buffer ready
  }

  // Epilogue per ISA 32-bit 16x16 C/D layout: elem r -> row M = r + 8*hi.
#pragma unroll
  for (int j = 0; j < 4; ++j) {
    const int n = (ntwg + ntw + j) * 16 + l;
    const float bv = bias[n];
#pragma unroll
    for (int i = 0; i < 2; ++i) {
      const int mb = m0 + i * 16 + hi * 8;
#pragma unroll
      for (int r = 0; r < 8; ++r)
        out[(size_t)(mb + r) * OUT_DIM + n] = c[i][j][r] + bv;
    }
  }
}

// Fallback (only if workspace can't hold the 32 MB packed weights).
__global__ __launch_bounds__(256) void eidetic_fallback(
    const float* __restrict__ x, const int* __restrict__ idx,
    const float* __restrict__ pt, const float* __restrict__ bias,
    float* __restrict__ out) {
  const int g = blockIdx.x * 256 + threadIdx.x;  // over B*OUT
  const int b = g >> 10, n = g & 1023;
  float acc = bias[n];
  for (int i = 0; i < IN_DIM; ++i) {
    const int q = idx[(size_t)b * IN_DIM + i];
    acc += x[(size_t)b * IN_DIM + i] *
           pt[(size_t)q * (IN_DIM * OUT_DIM) + (size_t)i * OUT_DIM + n];
  }
  out[g] = acc;
}

extern "C" void kernel_launch(void* const* d_in, const int* in_sizes, int n_in,
                              void* d_out, int out_size, void* d_ws, size_t ws_size,
                              hipStream_t stream) {
  const float* x    = (const float*)d_in[0];
  const int*   idx  = (const int*)d_in[1];
  const float* pt   = (const float*)d_in[2];
  const float* bias = (const float*)d_in[3];
  float* out = (float*)d_out;

  const size_t need = (size_t)Q_DIM * IN_DIM * OUT_DIM * sizeof(unsigned short); // 32 MB
  if (ws_size >= need) {
    unsigned short* bp = (unsigned short*)d_ws;
    eidetic_pack_bp<<<dim3(4096), dim3(256), 0, stream>>>(pt, bp);
    eidetic_wmma_gemm<<<dim3(OUT_DIM / 128, B_DIM / 128), dim3(256), 0, stream>>>(
        x, idx, bp, bias, out);
  } else {
    eidetic_fallback<<<dim3((B_DIM * OUT_DIM) / 256), dim3(256), 0, stream>>>(
        x, idx, pt, bias, out);
  }
}
